// MultiHeadAttention_58153857188517
// MI455X (gfx1250) — compile-verified, matrix operand order checked by reading, not run on hardware
//
#include <hip/hip_runtime.h>

// ---------- types ----------
typedef __attribute__((ext_vector_type(16))) __bf16 v16bf;
typedef __attribute__((ext_vector_type(8)))  __bf16 v8bf;
typedef __attribute__((ext_vector_type(4)))  __bf16 v4bf;
typedef __attribute__((ext_vector_type(8)))  float  v8f;
typedef __attribute__((ext_vector_type(4)))  float  fv4;
typedef __attribute__((ext_vector_type(4)))  unsigned int u32x4;
typedef __attribute__((ext_vector_type(8)))  int i32x8;
typedef __attribute__((ext_vector_type(4)))  int i32x4;

#define WMMA_BF16(A, B, C) \
  __builtin_amdgcn_wmma_f32_16x16x32_bf16(false, (A), false, (B), (short)0, (C), false, false)

static constexpr int Bb  = 4;
static constexpr int Tt  = 2048;
static constexpr int Dd  = 2048;
static constexpr int Hh  = 16;
static constexpr int HDd = 128;
static constexpr int Mrows = Bb * Tt;      // 8192
static constexpr int LDT  = 40;            // LDS tile row stride (32 data + 8 pad bf16)
static constexpr int TILE_ELEMS = 128 * LDT;       // one staged tile, bf16 elements
static constexpr unsigned TILE_BYTES = TILE_ELEMS * 2;  // 10240 B

// ---------- fragment loaders ----------
// CDNA5 16-bit A/B fragment layout (CDNA5 ISA): lane L holds row (L&15); lanes
// 0-15 carry K chunks [0..7] and [16..23], lanes 16-31 carry [8..15] and [24..31].
__device__ __forceinline__ v16bf frag_from_bf16(const __bf16* __restrict__ p, int ld, int lane) {
  const int idx = lane & 15, half = lane >> 4;
  const __bf16* q0 = p + (size_t)idx * ld + half * 8;
  v8bf c0 = *(const v8bf*)(q0);
  v8bf c1 = *(const v8bf*)(q0 + 16);
  return __builtin_shufflevector(c0, c1, 0, 1, 2, 3, 4, 5, 6, 7, 8, 9, 10, 11, 12, 13, 14, 15);
}

// ---------- TDM: stage one 128-row x 32-col bf16 tile (row stride 2048) to LDS ----------
// D# group1: data_size=1(2B); pad_enable; pad_interval=16 DWORDs (code 3) == one 32-elem row;
// pad_amount=4 DWORDs (code 3) -> LDS row stride = 40 bf16 = 80 B (16B aligned, bank-spread).
__device__ __forceinline__ void tdm_stage_tile(const __bf16* gsrc, unsigned lds_byte_off) {
  const unsigned long long ga = (unsigned long long)(uintptr_t)gsrc;
  u32x4 g0;
  g0[0] = 1u;                                                // count=1, user descriptor
  g0[1] = lds_byte_off;                                      // lds_addr
  g0[2] = (unsigned)ga;                                      // global_addr[31:0]
  g0[3] = (unsigned)((ga >> 32) & 0x01FFFFFFu) | (2u << 30); // addr[56:32] | type=2
  i32x8 g1;
  g1[0] = (int)0x06D10000u;     // data_size=1 | pad_en | pad_interval=3 | pad_amount=3
  g1[1] = (int)(2048u << 16);   // tensor_dim0 = 2048 (bits 79:48, low part)
  g1[2] = (int)(128u << 16);    // tensor_dim0 hi=0 | tensor_dim1 = 128
  g1[3] = (int)(32u << 16);     // tensor_dim1 hi=0 | tile_dim0 = 32
  g1[4] = (int)128;             // tile_dim1 = 128 | tile_dim2 = 0
  g1[5] = (int)2048;            // tensor_dim0_stride = 2048 (low 32)
  g1[6] = 0;                    // stride hi | tensor_dim1_stride lo
  g1[7] = 0;
  i32x4 gz4 = {0, 0, 0, 0};
  i32x8 gz8 = {0, 0, 0, 0, 0, 0, 0, 0};
  __builtin_amdgcn_tensor_load_to_lds(g0, g1, gz4, gz4, gz8, 0);
}

__device__ __forceinline__ unsigned lds_off_u32(const void* p) {
  return (unsigned)(uintptr_t)p;  // flat LDS address: low 32 bits = workgroup-relative offset
}

// ---------- kernel 0: elementwise f32 -> bf16 ----------
__global__ __launch_bounds__(256) void cvt_f32_bf16_kernel(const float* __restrict__ src,
                                                           __bf16* __restrict__ dst, int n4) {
  const int i = blockIdx.x * 256 + threadIdx.x;
  if (i < n4) {
    fv4 v = *(const fv4*)(src + (size_t)i * 4);
    v4bf o;
#pragma unroll
    for (int e = 0; e < 4; ++e) o[e] = (__bf16)v[e];
    *(v4bf*)(dst + (size_t)i * 4) = o;
  }
}

// ---------- staged 128x128 GEMM body (TDM double-buffered), wave tile 32x64 ----------
// No constant-initialized LDS-pointer arrays (lld cannot relocate addrspacecast
// aggregates): buffer bases are computed arithmetically from `smem` / its offset.
#define GEMM_STAGED_LOOP(Aptr, Bptr)                                                    \
  __shared__ __align__(16) __bf16 smem[4 * TILE_ELEMS];                                 \
  const unsigned smem_lds = lds_off_u32(smem);                                          \
  v8f acc[2][4];                                                                        \
  _Pragma("unroll") for (int mf = 0; mf < 2; ++mf)                                      \
  _Pragma("unroll") for (int nf = 0; nf < 4; ++nf)                                      \
  _Pragma("unroll") for (int g = 0; g < 8; ++g) acc[mf][nf][g] = 0.0f;                  \
  if (wv == 0) {                                                                        \
    tdm_stage_tile((Aptr) + (size_t)Mblk * Dd, smem_lds);                               \
    tdm_stage_tile((Bptr) + (size_t)Nblk * Dd, smem_lds + 2u * TILE_BYTES);             \
  }                                                                                     \
  __builtin_amdgcn_s_wait_tensorcnt(0);                                                 \
  __syncthreads();                                                                      \
  int buf = 0;                                                                          \
  for (int ks = 0; ks < Dd; ks += 32, buf ^= 1) {                                       \
    if (wv == 0 && ks + 32 < Dd) {                                                      \
      tdm_stage_tile((Aptr) + (size_t)Mblk * Dd + (ks + 32),                            \
                     smem_lds + (unsigned)(buf ^ 1) * TILE_BYTES);                      \
      tdm_stage_tile((Bptr) + (size_t)Nblk * Dd + (ks + 32),                            \
                     smem_lds + (unsigned)(2 + (buf ^ 1)) * TILE_BYTES);                \
    }                                                                                   \
    const __bf16* lxb = smem + buf * TILE_ELEMS;                                        \
    const __bf16* lwb = smem + (2 + buf) * TILE_ELEMS;                                  \
    v16bf a[2];                                                                         \
    _Pragma("unroll") for (int mf = 0; mf < 2; ++mf)                                    \
      a[mf] = frag_from_bf16(lxb + (wm * 32 + mf * 16) * LDT, LDT, lane);               \
    v16bf bfr[4];                                                                       \
    _Pragma("unroll") for (int nf = 0; nf < 4; ++nf)                                    \
      bfr[nf] = frag_from_bf16(lwb + (wn * 64 + nf * 16) * LDT, LDT, lane);             \
    _Pragma("unroll") for (int mf = 0; mf < 2; ++mf)                                    \
    _Pragma("unroll") for (int nf = 0; nf < 4; ++nf)                                    \
      acc[mf][nf] = WMMA_BF16(a[mf], bfr[nf], acc[mf][nf]);                             \
    __builtin_amdgcn_s_wait_tensorcnt(0);                                               \
    __syncthreads();                                                                    \
  }

// ---------- kernel 1: fused QKV projection, Y = Xb @ Wb^T (bf16 out) ----------
// grid = (D/128, M/128, 3); block = 256 (8 waves).
__global__ __launch_bounds__(256) void qkv_gemm_kernel(
    const __bf16* __restrict__ Xb,
    const __bf16* __restrict__ Wqb, const __bf16* __restrict__ Wkb, const __bf16* __restrict__ Wvb,
    __bf16* __restrict__ Qb, __bf16* __restrict__ Kb, __bf16* __restrict__ Vtb) {
  const int lane = threadIdx.x & 31;
  const int wv   = threadIdx.x >> 5;
  const int wm   = wv >> 1;  // 0..3
  const int wn   = wv & 1;   // 0..1
  const int idx  = lane & 15, half = lane >> 4;
  const int mode = blockIdx.z;
  const __bf16* Bm = (mode == 0) ? Wqb : (mode == 1) ? Wkb : Wvb;
  const int Mblk = blockIdx.y * 128;
  const int Nblk = blockIdx.x * 128;

  GEMM_STAGED_LOOP(Xb, Bm)

  // epilogue: scatter into head-major layouts (Q/K: [b,h,t,d], V: [b,h,d,t])
#pragma unroll
  for (int mf = 0; mf < 2; ++mf)
#pragma unroll
    for (int nf = 0; nf < 4; ++nf)
#pragma unroll
      for (int g = 0; g < 8; ++g) {
        const int m = Mblk + wm * 32 + mf * 16 + g + 8 * half;
        const int n = Nblk + wn * 64 + nf * 16 + idx;
        const __bf16 v = (__bf16)acc[mf][nf][g];
        const int bb = m >> 11, tt = m & (Tt - 1);
        const int hh = n >> 7,  dd = n & (HDd - 1);
        if (mode == 0)
          Qb[(((size_t)(bb * Hh + hh)) * Tt + tt) * HDd + dd] = v;
        else if (mode == 1)
          Kb[(((size_t)(bb * Hh + hh)) * Tt + tt) * HDd + dd] = v;
        else
          Vtb[(((size_t)(bb * Hh + hh)) * HDd + dd) * Tt + tt] = v;
      }
}

// ---------- kernel 2: flash attention ----------
// grid = (T/128, B*H); block = 256 (8 waves); wave = 16 Q rows x HD=128.
__global__ __launch_bounds__(256) void attn_kernel(
    const __bf16* __restrict__ Q, const __bf16* __restrict__ K,
    const __bf16* __restrict__ Vt, __bf16* __restrict__ CTX) {
  const int lane = threadIdx.x & 31;
  const int w    = threadIdx.x >> 5;
  const int idx  = lane & 15, half = lane >> 4;
  const int ibase_blk = blockIdx.x * 128;
  const int bh = blockIdx.y;
  const int b  = bh >> 4, h = bh & (Hh - 1);
  const size_t headQK = (size_t)bh * Tt * HDd;
  const __bf16* Qh = Q + headQK;
  const __bf16* Kh = K + headQK;
  const __bf16* Vh = Vt + headQK;  // [HDd][Tt]
  const int irow0 = ibase_blk + w * 16;

  __shared__ __align__(16) __bf16 plds_all[8 * 16 * 72];
  __bf16* plds = plds_all + w * (16 * 72);  // wave-private, padded stride 72

  v16bf qf[4];
#pragma unroll
  for (int kk = 0; kk < 4; ++kk)
    qf[kk] = frag_from_bf16(Qh + (size_t)irow0 * HDd + kk * 32, HDd, lane);

  v8f acc[8];
#pragma unroll
  for (int nf = 0; nf < 8; ++nf)
#pragma unroll
    for (int g = 0; g < 8; ++g) acc[nf][g] = 0.0f;
  float mrun[8], lrun[8];
#pragma unroll
  for (int g = 0; g < 8; ++g) { mrun[g] = -__builtin_inff(); lrun[g] = 0.0f; }

  const float scale = 0.08838834764831845f;  // 1/sqrt(128)
  const int jt_end = (ibase_blk + 125) >> 6;

  for (int jt = 0; jt <= jt_end; ++jt) {
    const int jbase = jt * 64;

    v8f s[4];
#pragma unroll
    for (int nf = 0; nf < 4; ++nf) {
#pragma unroll
      for (int g = 0; g < 8; ++g) s[nf][g] = 0.0f;
      const __bf16* kb = Kh + (size_t)(jbase + nf * 16) * HDd;
#pragma unroll
      for (int kk = 0; kk < 4; ++kk) {
        v16bf bf = frag_from_bf16(kb + kk * 32, HDd, lane);
        s[nf] = WMMA_BF16(qf[kk], bf, s[nf]);
      }
    }

    const bool tile_masked = (jbase + 63) > (irow0 - 2);
    float sv[4][8];
#pragma unroll
    for (int nf = 0; nf < 4; ++nf)
#pragma unroll
      for (int g = 0; g < 8; ++g) {
        float v = s[nf][g] * scale;
        if (tile_masked) {
          const int i = irow0 + g + 8 * half;
          const int j = jbase + nf * 16 + idx;
          if (j >= i - 1) v = -__builtin_inff();
        }
        sv[nf][g] = v;
      }

    float alpha[8];
#pragma unroll
    for (int g = 0; g < 8; ++g) {
      float v = fmaxf(fmaxf(sv[0][g], sv[1][g]), fmaxf(sv[2][g], sv[3][g]));
#pragma unroll
      for (int off = 1; off < 16; off <<= 1) v = fmaxf(v, __shfl_xor(v, off, 32));
      const float mn = fmaxf(mrun[g], v);
      alpha[g] = __expf(mrun[g] - mn);
      mrun[g] = mn;
    }

    float rs[8];
#pragma unroll
    for (int g = 0; g < 8; ++g) rs[g] = 0.0f;
#pragma unroll
    for (int nf = 0; nf < 4; ++nf)
#pragma unroll
      for (int g = 0; g < 8; ++g) {
        const float p = __expf(sv[nf][g] - mrun[g]);
        rs[g] += p;
        plds[(g + 8 * half) * 72 + nf * 16 + idx] = (__bf16)p;
      }
#pragma unroll
    for (int g = 0; g < 8; ++g) {
      float v = rs[g];
#pragma unroll
      for (int off = 1; off < 16; off <<= 1) v += __shfl_xor(v, off, 32);
      lrun[g] = lrun[g] * alpha[g] + v;
    }
#pragma unroll
    for (int nf = 0; nf < 8; ++nf)
#pragma unroll
      for (int g = 0; g < 8; ++g) acc[nf][g] *= alpha[g];

#pragma unroll
    for (int kc = 0; kc < 2; ++kc) {
      v16bf pa = frag_from_bf16(plds + kc * 32, 72, lane);
#pragma unroll
      for (int nf = 0; nf < 8; ++nf) {
        v16bf vb = frag_from_bf16(Vh + (size_t)(nf * 16) * Tt + jbase + kc * 32, Tt, lane);
        acc[nf] = WMMA_BF16(pa, vb, acc[nf]);
      }
    }
  }

#pragma unroll
  for (int nf = 0; nf < 8; ++nf)
#pragma unroll
    for (int g = 0; g < 8; ++g) {
      const int i   = irow0 + g + 8 * half;
      const int col = h * HDd + nf * 16 + idx;
      CTX[((size_t)(b * Tt + i)) * Dd + col] = (__bf16)(acc[nf][g] / lrun[g]);
    }
}

// ---------- kernel 3: output projection, Out = CTX @ Wob^T + bo (f32) ----------
__global__ __launch_bounds__(256) void out_gemm_kernel(
    const __bf16* __restrict__ CTX, const __bf16* __restrict__ Wob,
    const float* __restrict__ bo, float* __restrict__ Out) {
  const int lane = threadIdx.x & 31;
  const int wv   = threadIdx.x >> 5;
  const int wm   = wv >> 1;
  const int wn   = wv & 1;
  const int idx  = lane & 15, half = lane >> 4;
  const int Mblk = blockIdx.y * 128;
  const int Nblk = blockIdx.x * 128;

  GEMM_STAGED_LOOP(CTX, Wob)

#pragma unroll
  for (int nf = 0; nf < 4; ++nf) {
    const int n = Nblk + wn * 64 + nf * 16 + idx;
    const float bias = bo[n];
#pragma unroll
    for (int mf = 0; mf < 2; ++mf)
#pragma unroll
      for (int g = 0; g < 8; ++g) {
        const int m = Mblk + wm * 32 + mf * 16 + g + 8 * half;
        Out[(size_t)m * Dd + n] = acc[mf][nf][g] + bias;
      }
  }
}

// ---------- launch ----------
extern "C" void kernel_launch(void* const* d_in, const int* in_sizes, int n_in,
                              void* d_out, int out_size, void* d_ws, size_t ws_size,
                              hipStream_t stream) {
  (void)in_sizes; (void)n_in; (void)out_size; (void)ws_size;
  const float* X  = (const float*)d_in[0];
  const float* Wq = (const float*)d_in[1];
  const float* Wk = (const float*)d_in[2];
  const float* Wv = (const float*)d_in[3];
  const float* Wo = (const float*)d_in[4];
  const float* bo = (const float*)d_in[5];
  float* Out = (float*)d_out;

  const size_t NEL = (size_t)Bb * Tt * Dd;  // 16,777,216
  const size_t WEL = (size_t)Dd * Dd;       //  4,194,304
  __bf16* Qb  = (__bf16*)d_ws;              // [B,H,T,HD]
  __bf16* Kb  = Qb + NEL;                   // [B,H,T,HD]
  __bf16* Vtb = Kb + NEL;                   // [B,H,HD,T]
  __bf16* CTX = Vtb + NEL;                  // [B,T,D]
  __bf16* Xb  = CTX + NEL;                  // [M,D]
  __bf16* Wqb = Xb + NEL;
  __bf16* Wkb = Wqb + WEL;
  __bf16* Wvb = Wkb + WEL;
  __bf16* Wob = Wvb + WEL;

  dim3 blk(256, 1, 1);
  cvt_f32_bf16_kernel<<<(int)(NEL / 4 / 256), blk, 0, stream>>>(X, Xb, (int)(NEL / 4));
  cvt_f32_bf16_kernel<<<(int)(WEL / 4 / 256), blk, 0, stream>>>(Wq, Wqb, (int)(WEL / 4));
  cvt_f32_bf16_kernel<<<(int)(WEL / 4 / 256), blk, 0, stream>>>(Wk, Wkb, (int)(WEL / 4));
  cvt_f32_bf16_kernel<<<(int)(WEL / 4 / 256), blk, 0, stream>>>(Wv, Wvb, (int)(WEL / 4));
  cvt_f32_bf16_kernel<<<(int)(WEL / 4 / 256), blk, 0, stream>>>(Wo, Wob, (int)(WEL / 4));
  qkv_gemm_kernel<<<dim3(Dd / 128, Mrows / 128, 3), blk, 0, stream>>>(Xb, Wqb, Wkb, Wvb, Qb, Kb, Vtb);
  attn_kernel<<<dim3(Tt / 128, Bb * Hh), blk, 0, stream>>>(Qb, Kb, Vtb, CTX);
  out_gemm_kernel<<<dim3(Dd / 128, Mrows / 128), blk, 0, stream>>>(CTX, Wob, bo, Out);
}
